// OptimizedTraceGPT_1640677507617
// MI455X (gfx1250) — matrix operand order repaired and checked by reading, not compiled, not run on hardware
//
#include <hip/hip_runtime.h>
#include <hip/hip_bf16.h>

typedef __bf16 bf16;
typedef __attribute__((ext_vector_type(16))) __bf16 v16bf;
typedef __attribute__((ext_vector_type(8)))  float  v8f;

#define D_  256
#define H_  8
#define DH_ 32
#define L_  6
#define FF_ 1024
#define S_  1024
#define B_  8
#define T_  (B_*S_)
#define W_  64

// ---- CDNA5 async global->LDS path (guarded; falls back to VGPR staging) ----
#if defined(__HIP_DEVICE_COMPILE__) && __has_builtin(__builtin_amdgcn_global_load_async_to_lds_b128)
#define USE_ASYNC 1
#else
#define USE_ASYNC 0
#endif

typedef int v4i_ __attribute__((vector_size(16)));
typedef __attribute__((address_space(1))) v4i_ g_v4i;   // global
typedef __attribute__((address_space(3))) v4i_ l_v4i;   // LDS

__device__ __forceinline__ void cp_async16(void* lds, const void* g) {
#if USE_ASYNC
  __builtin_amdgcn_global_load_async_to_lds_b128(
      (g_v4i*)(unsigned long long)g,
      (l_v4i*)(unsigned int)(unsigned long long)lds,
      0, 0);
#else
  (void)lds; (void)g;
#endif
}
__device__ __forceinline__ void wait_async_all() {
#if USE_ASYNC
# if __has_builtin(__builtin_amdgcn_s_wait_asynccnt)
  __builtin_amdgcn_s_wait_asynccnt(0);
# else
  asm volatile("s_wait_asynccnt 0x0" ::: "memory");
# endif
#endif
}

// ---------------- device helpers ----------------
__device__ __forceinline__ float gelu_f(float x) {
  return 0.5f * x * (1.0f + erff(x * 0.70710678118654752f));
}

// GEMM epilogue flags
#define GF_BIAS  1
#define GF_ACCUM 2
#define GF_RESID 4
#define GF_GELU  8
#define GF_WF32  16
#define GF_WBF   32

// ---------------- WMMA GEMM ----------------
// C[M,N] (+=) A[M,K](bf16 row-major, optional per-batch row shift) @ B, where
// BwT is the WEIGHT PRE-TRANSPOSED to [N x K] bf16 (so LDS staging is pure b128).
// Block: 256 threads = 8 waves. Block tile 256(M) x 64(N), K-step 32,
// double-buffered LDS. Wave w owns rows [w*32, w*32+32): 2 A-frags x 4 B-frags
// -> 8 back-to-back v_wmma_f32_16x16x32_bf16 per K-step.
__global__ __launch_bounds__(256)
void gemm_bf16_kernel(const bf16* __restrict__ A, const bf16* __restrict__ BwT,
                      const float* __restrict__ bias, const float* __restrict__ resid,
                      float* __restrict__ Cf, bf16* __restrict__ Cbf,
                      int M, int N, int K, int flags, int shift, int batch_len)
{
  __shared__ __align__(16) bf16 As[2][256][40];  // padded rows (80B) to spread banks
  __shared__ __align__(16) bf16 Bt[2][64][40];   // Bt[n][k] tile (already transposed)

  const int tid  = threadIdx.x;
  const int wave = tid >> 5;
  const int lane = tid & 31;
  const int m0   = blockIdx.x * 256;
  const int n0   = blockIdx.y * 64;

  // ---- tile staging (async DMA if available, else VGPR round-trip) ----
  auto stageA = [&](int kt, int pb) {
    int r  = tid;               // one row per thread (256 rows)
    int gm = m0 + r;
    int srcRow = gm;
    bool valid = true;
    if (batch_len > 0) {
      int b = gm / batch_len;
      int s = gm - b * batch_len + shift;
      valid  = ((unsigned)s < (unsigned)batch_len);
      srcRow = b * batch_len + s;
    }
    bf16* dstRow = &As[pb][r][0];
    if (valid) {
      const bf16* src = A + (size_t)srcRow * K + kt;   // 64B contiguous
#if USE_ASYNC
      cp_async16(dstRow +  0, src +  0);
      cp_async16(dstRow +  8, src +  8);
      cp_async16(dstRow + 16, src + 16);
      cp_async16(dstRow + 24, src + 24);
#else
      const uint4* s4 = reinterpret_cast<const uint4*>(src);
      uint4* d4 = reinterpret_cast<uint4*>(dstRow);
      uint4 t0 = s4[0], t1 = s4[1], t2 = s4[2], t3 = s4[3];
      d4[0] = t0; d4[1] = t1; d4[2] = t2; d4[3] = t3;
#endif
    } else {
      uint4 z = make_uint4(0u,0u,0u,0u);
      uint4* d4 = reinterpret_cast<uint4*>(dstRow);
      d4[0] = z; d4[1] = z; d4[2] = z; d4[3] = z;
    }
  };
  auto stageB = [&](int kt, int pb) {
    int nl = tid >> 2;          // 0..63
    int ch = tid & 3;           // 16B chunk within the 64B row
    int n  = n0 + nl;
    bf16* dst = &Bt[pb][nl][ch * 8];
    if (n < N) {
      const bf16* src = BwT + (size_t)n * K + kt + ch * 8;
#if USE_ASYNC
      cp_async16(dst, src);
#else
      *reinterpret_cast<uint4*>(dst) = *reinterpret_cast<const uint4*>(src);
#endif
    } else {
      *reinterpret_cast<uint4*>(dst) = make_uint4(0u,0u,0u,0u);
    }
  };

  v8f acc[2][4];
#pragma unroll
  for (int rg = 0; rg < 2; ++rg)
#pragma unroll
    for (int j = 0; j < 4; ++j)
#pragma unroll
      for (int e = 0; e < 8; ++e) acc[rg][j][e] = 0.0f;

  // prologue: stage tile 0 into buffer 0
  stageA(0, 0);
  stageB(0, 0);
  wait_async_all();
  __syncthreads();

  int p = 0;
  for (int k0 = 0; k0 < K; k0 += 32) {
    int kn = k0 + 32;
    if (kn < K) {               // fire next tile into the other buffer
      stageA(kn, p ^ 1);
      stageB(kn, p ^ 1);
    }

    // ---- fragments (ISA 16-bit A 16x32 / B 32x16 layouts) ----
    union FragU { uint4 u[2]; v16bf v; };
    FragU fa[2], fb[4];
    {
      int half = lane >> 4;
#pragma unroll
      for (int rg = 0; rg < 2; ++rg) {
        int mrow = wave * 32 + rg * 16 + (lane & 15);
        fa[rg].u[0] = *reinterpret_cast<const uint4*>(&As[p][mrow][half * 8]);
        fa[rg].u[1] = *reinterpret_cast<const uint4*>(&As[p][mrow][16 + half * 8]);
      }
#pragma unroll
      for (int j = 0; j < 4; ++j) {
        int n  = j * 16 + (lane & 15);
        int kb = (lane >> 4) * 8;
        fb[j].u[0] = *reinterpret_cast<const uint4*>(&Bt[p][n][kb]);
        fb[j].u[1] = *reinterpret_cast<const uint4*>(&Bt[p][n][kb + 16]);
      }
    }
#pragma unroll
    for (int rg = 0; rg < 2; ++rg)
#pragma unroll
      for (int j = 0; j < 4; ++j)
        acc[rg][j] = __builtin_amdgcn_wmma_f32_16x16x32_bf16(
            false, fa[rg].v, false, fb[j].v, (short)0, acc[rg][j], false, false);

    wait_async_all();           // next tile landed
    __syncthreads();            // everyone done reading buffer p
    p ^= 1;
  }

  // ---- epilogue: lane (N=lane&15), VGPR r -> M=(lane>>4)*8+r ----
#pragma unroll
  for (int rg = 0; rg < 2; ++rg) {
#pragma unroll
    for (int j = 0; j < 4; ++j) {
      int col = n0 + j * 16 + (lane & 15);
      if (col >= N) continue;
#pragma unroll
      for (int r = 0; r < 8; ++r) {
        int row = m0 + wave * 32 + rg * 16 + (lane >> 4) * 8 + r;
        size_t idx = (size_t)row * N + col;
        float v = acc[rg][j][r];
        if (flags & GF_BIAS)  v += bias[col];
        if (flags & GF_ACCUM) v += Cf[idx];
        if (flags & GF_RESID) v += resid[idx];
        if (flags & GF_GELU)  v = gelu_f(v);
        if (flags & GF_WF32)  Cf[idx]  = v;
        if (flags & GF_WBF)   Cbf[idx] = (bf16)v;
      }
    }
  }
}

// ---------------- embedding + LN + RoPE (one wave per token) ----------------
__global__ __launch_bounds__(256)
void embed_kernel(const float* __restrict__ x, const float* __restrict__ w1,
                  const float* __restrict__ b1, const float* __restrict__ w2,
                  const float* __restrict__ b2, const float* __restrict__ g,
                  const float* __restrict__ bb, float* __restrict__ h)
{
  __shared__ float esh[8][128];
  __shared__ float hsh[8][256];
  int w = threadIdx.x >> 5, lane = threadIdx.x & 31;
  int tok = blockIdx.x * 8 + w;
  int s = tok % S_;
  float xv = x[tok];
#pragma unroll
  for (int r = 0; r < 4; ++r) { int j = r * 32 + lane; esh[w][j] = gelu_f(xv * w1[j] + b1[j]); }
  __syncthreads();
  float sum = 0.f, sq = 0.f, hv[8];
#pragma unroll
  for (int r = 0; r < 8; ++r) {
    int d = r * 32 + lane;
    float a = b2[d];
    for (int j = 0; j < 128; ++j) a += esh[w][j] * w2[j * D_ + d];
    hv[r] = a; sum += a; sq += a * a;
  }
  for (int off = 16; off; off >>= 1) { sum += __shfl_xor(sum, off, 32); sq += __shfl_xor(sq, off, 32); }
  float mean = sum * (1.f / D_);
  float var  = sq * (1.f / D_) - mean * mean;
  float rs   = rsqrtf(var + 1e-5f);
#pragma unroll
  for (int r = 0; r < 8; ++r) { int d = r * 32 + lane; hsh[w][d] = (hv[r] - mean) * rs * g[d] + bb[d]; }
  __syncthreads();
#pragma unroll
  for (int r = 0; r < 8; ++r) {
    int d = r * 32 + lane;
    int i2 = d & 127;
    float th = (float)s * __powf(10000.f, -((float)(2 * i2)) / 256.f);
    float rot = (d < 128) ? -hsh[w][d + 128] : hsh[w][d - 128];
    h[(size_t)tok * D_ + d] = hsh[w][d] * cosf(th) + rot * sinf(th);
  }
}

// ---------------- LayerNorm -> bf16 (one wave per token) ----------------
__global__ __launch_bounds__(256)
void ln_bf16_kernel(const float* __restrict__ in, const float* __restrict__ g,
                    const float* __restrict__ b, bf16* __restrict__ out)
{
  int wid = blockIdx.x * 8 + (threadIdx.x >> 5);
  int lane = threadIdx.x & 31;
  const float* row = in + (size_t)wid * D_;
  float v[8], s = 0.f, sq = 0.f;
#pragma unroll
  for (int r = 0; r < 8; ++r) { v[r] = row[r * 32 + lane]; s += v[r]; sq += v[r] * v[r]; }
  for (int off = 16; off; off >>= 1) { s += __shfl_xor(s, off, 32); sq += __shfl_xor(sq, off, 32); }
  float mean = s * (1.f / D_);
  float var  = sq * (1.f / D_) - mean * mean;
  float rs   = rsqrtf(var + 1e-5f);
#pragma unroll
  for (int r = 0; r < 8; ++r) {
    int d = r * 32 + lane;
    out[(size_t)wid * D_ + d] = (bf16)((v[r] - mean) * rs * g[d] + b[d]);
  }
}

// ---------------- sliding-window attention (one wave per (token, head)) ----------------
__global__ __launch_bounds__(256)
void attn_kernel(const float* __restrict__ q, const float* __restrict__ k,
                 const float* __restrict__ v, bf16* __restrict__ out)
{
  __shared__ float qsh[8][32];
  __shared__ float psh[8][96];
  int w = threadIdx.x >> 5, lane = threadIdx.x & 31;
  int wid  = blockIdx.x * 8 + w;       // 0..B*S*H-1
  int head = wid & (H_ - 1);
  int tok  = wid >> 3;
  int b = tok / S_, s = tok - b * S_;
  qsh[w][lane] = q[(size_t)tok * D_ + head * DH_ + lane];
  __syncthreads();
  int j0 = s - W_;
  float sc[3]; bool val[3];
  float m = -1e30f;
#pragma unroll
  for (int c = 0; c < 3; ++c) {
    int rel = c * 32 + lane;
    int j = j0 + rel;
    bool ok = (rel <= W_) && (j >= 0);
    val[c] = ok;
    float d = -1e30f;
    if (ok) {
      const float* krow = k + (size_t)(b * S_ + j) * D_ + head * DH_;
      d = 0.f;
      for (int t = 0; t < DH_; ++t) d += qsh[w][t] * krow[t];
      d *= 0.17677669529663687f;  // 1/sqrt(DH)
    }
    sc[c] = d; m = fmaxf(m, d);
  }
  for (int off = 16; off; off >>= 1) m = fmaxf(m, __shfl_xor(m, off, 32));
  float sum = 0.f;
#pragma unroll
  for (int c = 0; c < 3; ++c) {
    float p = val[c] ? __expf(sc[c] - m) : 0.f;
    psh[w][c * 32 + lane] = p;
    sum += p;
  }
  for (int off = 16; off; off >>= 1) sum += __shfl_xor(sum, off, 32);
  __syncthreads();
  float inv = 1.f / sum;
  float acc = 0.f;
  for (int rel = 0; rel <= W_; ++rel) {
    int j = j0 + rel;
    if (j < 0) continue;
    acc += psh[w][rel] * v[(size_t)(b * S_ + j) * D_ + head * DH_ + lane];
  }
  out[(size_t)tok * D_ + head * DH_ + lane] = (bf16)(acc * inv);
}

// ---------------- small elementwise kernels ----------------
__global__ void f2bf_kernel(const float* __restrict__ in, bf16* __restrict__ out, int n) {
  int i = blockIdx.x * 256 + threadIdx.x;
  if (i < n) out[i] = (bf16)in[i];
}
// fp32 [K x N] -> bf16 [N x K] (batched) : weight pre-transpose for the GEMM B path
__global__ void f2bfT_kernel(const float* __restrict__ in, bf16* __restrict__ out,
                             int K, int N, int total) {
  int i = blockIdx.x * 256 + threadIdx.x;
  if (i >= total) return;
  int per = K * N;
  int b = i / per, r = i - b * per;
  int n = r / K, k = r - n * K;
  out[i] = (bf16)in[(size_t)b * per + (size_t)k * N + n];
}
__global__ void gelu2bf_kernel(const float* __restrict__ in, bf16* __restrict__ out, int n) {
  int i = blockIdx.x * 256 + threadIdx.x;
  if (i < n) out[i] = (bf16)gelu_f(in[i]);
}
__global__ void sigmoid_kernel(const float* __restrict__ in, float* __restrict__ out, int n) {
  int i = blockIdx.x * 256 + threadIdx.x;
  if (i < n) out[i] = 1.f / (1.f + __expf(-in[i]));
}
// conv weight (O,I,T) -> bf16 [T][O][I]  (i.e. per-tap, pre-transposed [N=O][K=I])
__global__ void convw_t_kernel(const float* __restrict__ in, bf16* __restrict__ out, int taps) {
  int i = blockIdx.x * 256 + threadIdx.x;
  int total = D_ * D_ * taps;
  if (i >= total) return;
  int t = i / (D_ * D_);
  int rem = i - t * D_ * D_;
  int o = rem / D_, ii = rem % D_;
  out[i] = (bf16)in[(o * D_ + ii) * taps + t];
}

// ---------------- gelu + mean over sequence: tf[b,o] ----------------
__global__ void meanpool_kernel(const float* __restrict__ c2, float* __restrict__ tf) {
  int i = blockIdx.x * 256 + threadIdx.x;  // 0..2047
  int b = i >> 8, o = i & 255;
  const float* base = c2 + (size_t)b * S_ * D_ + o;
  float a = 0.f;
  for (int s = 0; s < S_; ++s) a += gelu_f(base[(size_t)s * D_]);
  tf[i] = a * (1.f / S_);
}

// ---------------- classifier head (one block) ----------------
__global__ __launch_bounds__(256)
void cls_kernel(const float* __restrict__ tf, const float* __restrict__ w1,
                const float* __restrict__ b1, const float* __restrict__ w2,
                const float* __restrict__ b2, const float* __restrict__ w3,
                const float* __restrict__ b3, float* __restrict__ out)
{
  __shared__ float s1[8][256];
  __shared__ float s2[8][128];
  int t = threadIdx.x;
  for (int r = 0; r < 8; ++r) {
    float a = b1[t];
    for (int kk = 0; kk < 256; ++kk) a += tf[r * 256 + kk] * w1[kk * 256 + t];
    s1[r][t] = gelu_f(a);
  }
  __syncthreads();
  if (t < 128) {
    for (int r = 0; r < 8; ++r) {
      float a = b2[t];
      for (int kk = 0; kk < 256; ++kk) a += s1[r][kk] * w2[kk * 128 + t];
      s2[r][t] = gelu_f(a);
    }
  }
  __syncthreads();
  if (t < 8) {
    float a = b3[0];
    for (int kk = 0; kk < 128; ++kk) a += s2[t][kk] * w3[kk];
    out[t] = 1.f / (1.f + __expf(-a));
  }
}

// ---------------- host side ----------------
static void gemm(hipStream_t st, const bf16* A, const bf16* BwT, const float* bias,
                 const float* resid, float* Cf, bf16* Cbf, int M, int N, int K,
                 int flags, int shift = 0, int blen = 0)
{
  dim3 grid(M / 256, (N + 63) / 64);
  gemm_bf16_kernel<<<grid, 256, 0, st>>>(A, BwT, bias, resid, Cf, Cbf, M, N, K, flags, shift, blen);
}
static void cvtT(hipStream_t st, const float* in, bf16* out, int K, int N, int batches) {
  int total = K * N * batches;
  f2bfT_kernel<<<(total + 255) / 256, 256, 0, st>>>(in, out, K, N, total);
}

// input indices (setup_inputs dict order, params flattened in insertion order)
enum {
  IN_X = 0, IN_EMB_W1, IN_EMB_B1, IN_EMB_W2, IN_EMB_B2, IN_EMB_LN_G, IN_EMB_LN_B,
  IN_LN1_G, IN_LN1_B, IN_LN2_G, IN_LN2_B,
  IN_WQ, IN_BQ, IN_WK, IN_BK, IN_WV, IN_BV, IN_WO, IN_BO,
  IN_FFN_W1, IN_FFN_B1, IN_FFN_W2, IN_FFN_B2,
  IN_A_WQ, IN_A_BQ, IN_A_WK, IN_A_BK, IN_A_WV, IN_A_BV, IN_A_WO, IN_A_BO,
  IN_AH_LN_G, IN_AH_LN_B, IN_AH_W1, IN_AH_B1, IN_AH_W2, IN_AH_B2, IN_AH_W3, IN_AH_B3,
  IN_CONV_W1, IN_CONV_B1, IN_CONV_W2, IN_CONV_B2,
  IN_CLS_W1, IN_CLS_B1, IN_CLS_W2, IN_CLS_B2, IN_CLS_W3, IN_CLS_B3
};

extern "C" void kernel_launch(void* const* d_in, const int* in_sizes, int n_in,
                              void* d_out, int out_size, void* d_ws, size_t ws_size,
                              hipStream_t stream)
{
  (void)in_sizes; (void)n_in; (void)out_size; (void)ws_size;
  const float* F[49];
  for (int i = 0; i < 49; ++i) F[i] = (const float*)d_in[i];
  float* out = (float*)d_out;

  char* ws = (char*)d_ws;
  size_t off = 0;
  auto alloc = [&](size_t bytes) -> void* {
    void* p = ws + off;
    off += (bytes + 255) & ~(size_t)255;
    return p;
  };

  float* h      = (float*)alloc((size_t)T_ * D_ * 4);
  float* qb     = (float*)alloc((size_t)T_ * D_ * 4);
  float* kb     = (float*)alloc((size_t)T_ * D_ * 4);
  float* vb     = (float*)alloc((size_t)T_ * D_ * 4);
  float* af     = (float*)alloc((size_t)T_ * D_ * 4);
  float* c1     = (float*)alloc((size_t)T_ * D_ * 4);
  float* c2     = (float*)alloc((size_t)T_ * D_ * 4);
  float* logits = (float*)alloc((size_t)T_ * 4);
  float* tf     = (float*)alloc((size_t)B_ * D_ * 4);
  bf16* nh_bf   = (bf16*)alloc((size_t)T_ * D_ * 2);
  bf16* ao_bf   = (bf16*)alloc((size_t)T_ * D_ * 2);
  bf16* hbf     = (bf16*)alloc((size_t)T_ * D_ * 2);
  bf16* c1bf    = (bf16*)alloc((size_t)T_ * D_ * 2);
  bf16* mid_bf  = (bf16*)alloc((size_t)T_ * FF_ * 2);
  bf16* t2_bf   = (bf16*)alloc((size_t)T_ * 128 * 2);
  // bf16 weights (all pre-transposed to [N x K])
  bf16* wq_bf   = (bf16*)alloc((size_t)L_ * D_ * D_ * 2);
  bf16* wk_bf   = (bf16*)alloc((size_t)L_ * D_ * D_ * 2);
  bf16* wv_bf   = (bf16*)alloc((size_t)L_ * D_ * D_ * 2);
  bf16* wo_bf   = (bf16*)alloc((size_t)L_ * D_ * D_ * 2);
  bf16* f1_bf   = (bf16*)alloc((size_t)L_ * D_ * FF_ * 2);
  bf16* f2_bf   = (bf16*)alloc((size_t)L_ * FF_ * D_ * 2);
  bf16* awq_bf  = (bf16*)alloc((size_t)D_ * D_ * 2);
  bf16* awk_bf  = (bf16*)alloc((size_t)D_ * D_ * 2);
  bf16* awv_bf  = (bf16*)alloc((size_t)D_ * D_ * 2);
  bf16* awo_bf  = (bf16*)alloc((size_t)D_ * D_ * 2);
  bf16* ahw1_bf = (bf16*)alloc((size_t)D_ * D_ * 2);
  bf16* ahw2_bf = (bf16*)alloc((size_t)D_ * 128 * 2);
  bf16* ahw3_bf = (bf16*)alloc((size_t)128 * 2);
  bf16* w1t_bf  = (bf16*)alloc((size_t)3 * D_ * D_ * 2);
  bf16* w2t_bf  = (bf16*)alloc((size_t)5 * D_ * D_ * 2);

  // ---- convert weights to bf16, transposed to [N x K] ----
  cvtT(stream, F[IN_WQ],     wq_bf,  D_,  D_,  L_);
  cvtT(stream, F[IN_WK],     wk_bf,  D_,  D_,  L_);
  cvtT(stream, F[IN_WV],     wv_bf,  D_,  D_,  L_);
  cvtT(stream, F[IN_WO],     wo_bf,  D_,  D_,  L_);
  cvtT(stream, F[IN_FFN_W1], f1_bf,  D_,  FF_, L_);
  cvtT(stream, F[IN_FFN_W2], f2_bf,  FF_, D_,  L_);
  cvtT(stream, F[IN_A_WQ],   awq_bf, D_,  D_,  1);
  cvtT(stream, F[IN_A_WK],   awk_bf, D_,  D_,  1);
  cvtT(stream, F[IN_A_WV],   awv_bf, D_,  D_,  1);
  cvtT(stream, F[IN_A_WO],   awo_bf, D_,  D_,  1);
  cvtT(stream, F[IN_AH_W1],  ahw1_bf, D_, D_,  1);
  cvtT(stream, F[IN_AH_W2],  ahw2_bf, D_, 128, 1);
  cvtT(stream, F[IN_AH_W3],  ahw3_bf, 128, 1,  1);
  convw_t_kernel<<<(3 * D_ * D_ + 255) / 256, 256, 0, stream>>>(F[IN_CONV_W1], w1t_bf, 3);
  convw_t_kernel<<<(5 * D_ * D_ + 255) / 256, 256, 0, stream>>>(F[IN_CONV_W2], w2t_bf, 5);

  // ---- embedding + LN + RoPE ----
  embed_kernel<<<T_ / 8, 256, 0, stream>>>(F[IN_X], F[IN_EMB_W1], F[IN_EMB_B1],
                                           F[IN_EMB_W2], F[IN_EMB_B2],
                                           F[IN_EMB_LN_G], F[IN_EMB_LN_B], h);

  // ---- transformer layers ----
  for (int l = 0; l < L_; ++l) {
    const size_t wOff = (size_t)l * D_ * D_;
    const size_t fOff = (size_t)l * D_ * FF_;
    ln_bf16_kernel<<<T_ / 8, 256, 0, stream>>>(h, F[IN_LN1_G] + l * D_, F[IN_LN1_B] + l * D_, nh_bf);
    gemm(stream, nh_bf, wq_bf + wOff, F[IN_BQ] + l * D_, nullptr, qb, nullptr, T_, D_, D_, GF_BIAS | GF_WF32);
    gemm(stream, nh_bf, wk_bf + wOff, F[IN_BK] + l * D_, nullptr, kb, nullptr, T_, D_, D_, GF_BIAS | GF_WF32);
    gemm(stream, nh_bf, wv_bf + wOff, F[IN_BV] + l * D_, nullptr, vb, nullptr, T_, D_, D_, GF_BIAS | GF_WF32);
    attn_kernel<<<(T_ * H_) / 8, 256, 0, stream>>>(qb, kb, vb, ao_bf);
    gemm(stream, ao_bf, wo_bf + wOff, F[IN_BO] + l * D_, h, h, nullptr, T_, D_, D_, GF_BIAS | GF_RESID | GF_WF32);
    ln_bf16_kernel<<<T_ / 8, 256, 0, stream>>>(h, F[IN_LN2_G] + l * D_, F[IN_LN2_B] + l * D_, nh_bf);
    gemm(stream, nh_bf, f1_bf + fOff, F[IN_FFN_B1] + l * FF_, nullptr, nullptr, mid_bf, T_, FF_, D_,
         GF_BIAS | GF_GELU | GF_WBF);
    gemm(stream, mid_bf, f2_bf + fOff, F[IN_FFN_B2] + l * D_, h, h, nullptr, T_, D_, FF_,
         GF_BIAS | GF_RESID | GF_WF32);
  }

  // ---- extra attention head on raw h: af = MHA(h) + h ----
  f2bf_kernel<<<(T_ * D_) / 256, 256, 0, stream>>>(h, hbf, T_ * D_);
  gemm(stream, hbf, awq_bf, F[IN_A_BQ], nullptr, qb, nullptr, T_, D_, D_, GF_BIAS | GF_WF32);
  gemm(stream, hbf, awk_bf, F[IN_A_BK], nullptr, kb, nullptr, T_, D_, D_, GF_BIAS | GF_WF32);
  gemm(stream, hbf, awv_bf, F[IN_A_BV], nullptr, vb, nullptr, T_, D_, D_, GF_BIAS | GF_WF32);
  attn_kernel<<<(T_ * H_) / 8, 256, 0, stream>>>(qb, kb, vb, ao_bf);
  gemm(stream, ao_bf, awo_bf, F[IN_A_BO], h, af, nullptr, T_, D_, D_, GF_BIAS | GF_RESID | GF_WF32);

  // ---- point-score head ----
  ln_bf16_kernel<<<T_ / 8, 256, 0, stream>>>(af, F[IN_AH_LN_G], F[IN_AH_LN_B], nh_bf);
  gemm(stream, nh_bf, ahw1_bf, F[IN_AH_B1], nullptr, nullptr, mid_bf, T_, D_, D_, GF_BIAS | GF_GELU | GF_WBF);
  gemm(stream, mid_bf, ahw2_bf, F[IN_AH_B2], nullptr, nullptr, t2_bf, T_, 128, D_, GF_BIAS | GF_GELU | GF_WBF);
  gemm(stream, t2_bf, ahw3_bf, F[IN_AH_B3], nullptr, logits, nullptr, T_, 1, 128, GF_BIAS | GF_WF32);
  sigmoid_kernel<<<T_ / 256, 256, 0, stream>>>(logits, out, T_);

  // ---- conv branch (shifted WMMA GEMMs over the sequence axis) ----
  for (int t = 0; t < 3; ++t)
    gemm(stream, hbf, w1t_bf + (size_t)t * D_ * D_, F[IN_CONV_B1], nullptr, c1, nullptr, T_, D_, D_,
         (t == 0 ? (GF_BIAS | GF_WF32) : (GF_ACCUM | GF_WF32)), t - 1, S_);
  gelu2bf_kernel<<<(T_ * D_) / 256, 256, 0, stream>>>(c1, c1bf, T_ * D_);
  for (int t = 0; t < 5; ++t)
    gemm(stream, c1bf, w2t_bf + (size_t)t * D_ * D_, F[IN_CONV_B2], nullptr, c2, nullptr, T_, D_, D_,
         (t == 0 ? (GF_BIAS | GF_WF32) : (GF_ACCUM | GF_WF32)), t - 2, S_);
  meanpool_kernel<<<(B_ * D_) / 256, 256, 0, stream>>>(c2, tf);
  cls_kernel<<<1, 256, 0, stream>>>(tf, F[IN_CLS_W1], F[IN_CLS_B1], F[IN_CLS_W2], F[IN_CLS_B2],
                                    F[IN_CLS_W3], F[IN_CLS_B3], out + T_);
}
